// CryptoInformer_5858335391974
// MI455X (gfx1250) — matrix operand order repaired and checked by reading, not compiled, not run on hardware
//
#include <hip/hip_runtime.h>
#include <hip/hip_bf16.h>
#include <math.h>

typedef _Float16 f16;
typedef __attribute__((ext_vector_type(16))) _Float16 v16h;
typedef __attribute__((ext_vector_type(8)))  _Float16 v8h;
typedef __attribute__((ext_vector_type(8)))  float    v8f;

#define B_    2
#define S_    4095
#define FIN   32
#define D_    512
#define H_    8
#define E_    64
#define DFF_  2048
#define L_    4096            // S_ + 1 (cls token)
#define MR_   (B_ * L_)       // 8192 rows
#define U_    45              // 5 * ceil(ln(4096))
#define NL_   2
#define EPS_  1e-5f

#define LDS_STRIDE 40         // halves per LDS tile row (32 data + 8 pad, 80B)

// ---------------------------------------------------------------- async copy
// global -> LDS 16-byte async DMA (ASYNCcnt path); LDS offset = low 32 bits of
// the flat shared-pointer address (aperture scheme keeps offset in addr[31:0]).
__device__ __forceinline__ void async_ld16(void* lds_ptr, const f16* gptr)
{
    unsigned loff = (unsigned)(uintptr_t)lds_ptr;
    asm volatile("global_load_async_to_lds_b128 %0, %1, off"
                 :: "v"(loff), "v"(gptr)
                 : "memory");
}

__device__ __forceinline__ void wait_async0()
{
    asm volatile("s_wait_asynccnt 0x0" ::: "memory");
}

// ---------------------------------------------------------------- embedding
__global__ void embed_kernel(const float* __restrict__ x, const float* __restrict__ We,
                             const float* __restrict__ be, const float* __restrict__ cls,
                             float* __restrict__ h)
{
    int row = blockIdx.x;            // 0 .. MR_-1
    int b = row / L_;
    int t = row % L_;
    const float LN1E4 = 9.210340371976184f;  // ln(10000)
    for (int d = threadIdx.x; d < D_; d += blockDim.x) {
        float v;
        if (t == 0) {
            v = cls[d];
        } else {
            const float* xr = x + ((size_t)b * S_ + (size_t)(t - 1)) * FIN;
            const float* wr = We + (size_t)d * FIN;
            float acc = 0.f;
            #pragma unroll
            for (int i = 0; i < FIN; ++i) acc += xr[i] * wr[i];
            v = acc + be[d];
        }
        int   i2  = d >> 1;
        float div = expf(-((float)(2 * i2)) * LN1E4 / (float)D_);
        float ang = (float)t * div;
        v += (d & 1) ? cosf(ang) : sinf(ang);
        h[(size_t)row * D_ + d] = v;
    }
}

// ---------------------------------------------------------------- f32 -> f16
__global__ void cvt16_kernel(const float* __restrict__ src, f16* __restrict__ dst, int n)
{
    int i = blockIdx.x * blockDim.x + threadIdx.x;
    if (i < n) dst[i] = (f16)src[i];
}

// ---------------------------------------------------------------- WMMA GEMM
// C[M,N] = A[M,K] * W[N,K]^T + bias, optional relu; f32 and/or f16 outputs.
// 256 threads = 8 waves (2 in M x 4 in N). Wave tile 64x32 = 4x2 subtiles.
// Double-buffered async global->LDS tile DMA, ds_load_b128 fragment reads.
__global__ __launch_bounds__(256)
void gemm_wmma_kernel(const f16* __restrict__ A, const f16* __restrict__ W,
                      const float* __restrict__ bias,
                      float* __restrict__ C, f16* __restrict__ C16,
                      int M, int N, int K, int relu_flag)
{
    __shared__ f16 ldsA[2][128 * LDS_STRIDE];   // 2 x 10240 B
    __shared__ f16 ldsB[2][128 * LDS_STRIDE];

    const int tid   = threadIdx.x;
    const int lane  = tid & 31;
    const int wid   = tid >> 5;
    const int waveM = wid & 1;
    const int waveN = wid >> 1;
    const int blockM = blockIdx.y * 128;
    const int blockN = blockIdx.x * 128;
    const int r  = lane & 15;
    const int hs = lane >> 4;

    // cooperative tile DMA: 512 chunks of 16B per matrix, 2 per thread each
    const int row0 = tid >> 1;                 // 0..127
    const int q0   = (tid & 1) * 2;            // 0 or 2

    v8f acc[4][2];
    const v8f vzero = {0.f,0.f,0.f,0.f,0.f,0.f,0.f,0.f};
    #pragma unroll
    for (int mt = 0; mt < 4; ++mt)
        #pragma unroll
        for (int nt = 0; nt < 2; ++nt) acc[mt][nt] = vzero;

    const f16* gA = A + (size_t)(blockM + row0) * K;
    const f16* gW = W + (size_t)(blockN + row0) * K;

    // prologue: tile 0 -> buffer 0
    #pragma unroll
    for (int c = 0; c < 2; ++c) {
        int qq = q0 + c;
        async_ld16(&ldsA[0][row0 * LDS_STRIDE + qq * 8], gA + qq * 8);
        async_ld16(&ldsB[0][row0 * LDS_STRIDE + qq * 8], gW + qq * 8);
    }

    const int nT = K >> 5;
    for (int it = 0; it < nT; ++it) {
        const int buf = it & 1;
        wait_async0();
        __syncthreads();          // tile(it) visible to all waves; prev reads done
        if (it + 1 < nT) {
            const int kk = (it + 1) << 5;
            #pragma unroll
            for (int c = 0; c < 2; ++c) {
                int qq = q0 + c;
                async_ld16(&ldsA[buf ^ 1][row0 * LDS_STRIDE + qq * 8], gA + kk + qq * 8);
                async_ld16(&ldsB[buf ^ 1][row0 * LDS_STRIDE + qq * 8], gW + kk + qq * 8);
            }
        }

        v16h afrag[4];
        #pragma unroll
        for (int mt = 0; mt < 4; ++mt) {
            const f16* ap = &ldsA[buf][(waveM * 64 + mt * 16 + r) * LDS_STRIDE + hs * 8];
            v8h lo = *(const v8h*)ap;
            v8h hi = *(const v8h*)(ap + 16);
            afrag[mt] = __builtin_shufflevector(lo, hi,
                0,1,2,3,4,5,6,7,8,9,10,11,12,13,14,15);
        }
        v16h bfrag[2];
        #pragma unroll
        for (int nt = 0; nt < 2; ++nt) {
            const f16* wp = &ldsB[buf][(waveN * 32 + nt * 16 + r) * LDS_STRIDE + hs * 8];
            v8h lo = *(const v8h*)wp;
            v8h hi = *(const v8h*)(wp + 16);
            bfrag[nt] = __builtin_shufflevector(lo, hi,
                0,1,2,3,4,5,6,7,8,9,10,11,12,13,14,15);
        }
        #pragma unroll
        for (int mt = 0; mt < 4; ++mt)
            #pragma unroll
            for (int nt = 0; nt < 2; ++nt)
                acc[mt][nt] = __builtin_amdgcn_wmma_f32_16x16x32_f16(
                    false, afrag[mt], false, bfrag[nt],
                    (short)0, acc[mt][nt], false, false);
    }

    const int m0  = blockM + waveM * 64;
    const int n0  = blockN + waveN * 32;
    const int col   = lane & 15;
    const int rbase = (lane >> 4) * 8;
    #pragma unroll
    for (int mt = 0; mt < 4; ++mt) {
        #pragma unroll
        for (int nt = 0; nt < 2; ++nt) {
            int   n    = n0 + nt * 16 + col;
            float bval = bias ? bias[n] : 0.f;
            #pragma unroll
            for (int vv = 0; vv < 8; ++vv) {
                int   m   = m0 + mt * 16 + rbase + vv;
                float val = acc[mt][nt][vv] + bval;
                if (relu_flag && val < 0.f) val = 0.f;
                if (C)   C[(size_t)m * N + n]   = val;
                if (C16) C16[(size_t)m * N + n] = (f16)val;
            }
        }
    }
}

// ---------------------------------------------------------------- prob-sparse M
__device__ __forceinline__ unsigned hash_u32(unsigned x)
{
    x ^= x >> 16; x *= 0x7feb352dU;
    x ^= x >> 15; x *= 0x846ca68bU;
    x ^= x >> 16; return x;
}

// one wave per (b,h,l): M = max_j(q . K_sample_j) - sum_j / L
__global__ void mscore_kernel(const float* __restrict__ q, const float* __restrict__ k,
                              float* __restrict__ Mout, int layer)
{
    int warp = (blockIdx.x * blockDim.x + threadIdx.x) >> 5;
    int lane = threadIdx.x & 31;
    if (warp >= B_ * H_ * L_) return;
    int l  = warp & (L_ - 1);
    int bh = warp >> 12;
    int b  = bh >> 3;
    int hh = bh & 7;
    const float* qrow = q + ((size_t)(b * L_ + l)) * D_ + hh * E_;
    float q0 = qrow[lane * 2], q1 = qrow[lane * 2 + 1];
    float mx = -INFINITY, sm = 0.f;
    for (int j = 0; j < U_; ++j) {
        unsigned kidx = hash_u32(((unsigned)(l * 64 + j)) ^ ((unsigned)layer * 0x9E3779B9U)) & (L_ - 1);
        const float* krow = k + ((size_t)(b * L_ + (int)kidx)) * D_ + hh * E_;
        float p = q0 * krow[lane * 2] + q1 * krow[lane * 2 + 1];
        #pragma unroll
        for (int off = 16; off > 0; off >>= 1) p += __shfl_xor(p, off, 32);
        mx = fmaxf(mx, p);
        sm += p;
    }
    if (lane == 0) Mout[warp] = mx - sm / (float)L_;
}

// ---------------------------------------------------------------- top-k (u=45)
__global__ void topk_kernel(const float* __restrict__ Mv, int* __restrict__ top)
{
    __shared__ float sv[L_];
    __shared__ float rv[256];
    __shared__ int   ri[256];
    int bh = blockIdx.x;
    const float* Mrow = Mv + (size_t)bh * L_;
    for (int i = threadIdx.x; i < L_; i += 256) sv[i] = Mrow[i];
    __syncthreads();
    for (int it = 0; it < U_; ++it) {
        float best = -INFINITY; int bi = L_;
        for (int i = threadIdx.x; i < L_; i += 256) {
            float v = sv[i];
            if (v > best || (v == best && i < bi)) { best = v; bi = i; }
        }
        rv[threadIdx.x] = best; ri[threadIdx.x] = bi;
        __syncthreads();
        for (int s = 128; s > 0; s >>= 1) {
            if (threadIdx.x < s) {
                float v2 = rv[threadIdx.x + s]; int i2 = ri[threadIdx.x + s];
                if (v2 > rv[threadIdx.x] ||
                    (v2 == rv[threadIdx.x] && i2 < ri[threadIdx.x])) {
                    rv[threadIdx.x] = v2; ri[threadIdx.x] = i2;
                }
            }
            __syncthreads();
        }
        if (threadIdx.x == 0) { top[(size_t)bh * U_ + it] = ri[0]; sv[ri[0]] = -INFINITY; }
        __syncthreads();
    }
}

// ---------------------------------------------------------------- v mean + ctx init
__global__ void vmean_kernel(const float* __restrict__ v, float* __restrict__ vmean)
{
    int idx = blockIdx.x * blockDim.x + threadIdx.x;  // b*D + d
    if (idx >= B_ * D_) return;
    int b = idx / D_, d = idx % D_;
    float s = 0.f;
    for (int l = 0; l < L_; ++l) s += v[((size_t)(b * L_ + l)) * D_ + d];
    vmean[idx] = s / (float)L_;
}

__global__ void ctx_init_kernel(const float* __restrict__ vmean, float* __restrict__ ctx)
{
    size_t i = (size_t)blockIdx.x * blockDim.x + threadIdx.x;
    if (i >= (size_t)MR_ * D_) return;
    int d = (int)(i % D_);
    int b = (int)(i / ((size_t)L_ * D_));
    ctx[i] = vmean[b * D_ + d];
}

// ---------------------------------------------------------------- sparse attention
// one block (128 thr) per (b,h,ui): full softmax over L keys for selected query
__global__ void sparse_attn_kernel(const float* __restrict__ q, const float* __restrict__ k,
                                   const float* __restrict__ v, const int* __restrict__ top,
                                   float* __restrict__ ctx)
{
    __shared__ float sc[L_];
    __shared__ float qsh[E_];
    __shared__ float red[128];
    int blk = blockIdx.x;
    int ui  = blk % U_;
    int bh  = blk / U_;
    int b   = bh >> 3, hh = bh & 7;
    int t   = top[(size_t)bh * U_ + ui];
    const int tid = threadIdx.x;
    if (tid < E_) qsh[tid] = q[((size_t)(b * L_ + t)) * D_ + hh * E_ + tid];
    __syncthreads();

    float lm = -INFINITY;
    for (int kk = tid; kk < L_; kk += 128) {
        const float* krow = k + ((size_t)(b * L_ + kk)) * D_ + hh * E_;
        float acc = 0.f;
        #pragma unroll
        for (int e = 0; e < E_; ++e) acc += qsh[e] * krow[e];
        acc *= 0.125f;  // 1/sqrt(64)
        sc[kk] = acc;
        lm = fmaxf(lm, acc);
    }
    red[tid] = lm; __syncthreads();
    for (int s = 64; s > 0; s >>= 1) {
        if (tid < s) red[tid] = fmaxf(red[tid], red[tid + s]);
        __syncthreads();
    }
    float mx = red[0]; __syncthreads();

    float ls = 0.f;
    for (int kk = tid; kk < L_; kk += 128) {
        float e = expf(sc[kk] - mx); sc[kk] = e; ls += e;
    }
    red[tid] = ls; __syncthreads();
    for (int s = 64; s > 0; s >>= 1) {
        if (tid < s) red[tid] += red[tid + s];
        __syncthreads();
    }
    float inv = 1.f / red[0];
    __syncthreads();

    if (tid < E_) {
        float acc = 0.f;
        for (int kk = 0; kk < L_; ++kk)
            acc += sc[kk] * v[((size_t)(b * L_ + kk)) * D_ + hh * E_ + tid];
        ctx[((size_t)(b * L_ + t)) * D_ + hh * E_ + tid] = acc * inv;
    }
}

// ---------------------------------------------------------------- residual + LN
__global__ void add_ln_kernel(const float* __restrict__ xin, const float* __restrict__ res,
                              const float* __restrict__ g, const float* __restrict__ bb,
                              float* __restrict__ out)
{
    __shared__ float red[256];
    int row = blockIdx.x;
    const float* xr = xin + (size_t)row * D_;
    const float* rr = res + (size_t)row * D_;
    float vals[2];
    float s = 0.f;
    #pragma unroll
    for (int i = 0; i < 2; ++i) {
        vals[i] = xr[threadIdx.x + i * 256] + rr[threadIdx.x + i * 256];
        s += vals[i];
    }
    red[threadIdx.x] = s; __syncthreads();
    for (int st = 128; st > 0; st >>= 1) {
        if (threadIdx.x < st) red[threadIdx.x] += red[threadIdx.x + st];
        __syncthreads();
    }
    float mu = red[0] / (float)D_; __syncthreads();
    float vs = 0.f;
    #pragma unroll
    for (int i = 0; i < 2; ++i) { float d = vals[i] - mu; vs += d * d; }
    red[threadIdx.x] = vs; __syncthreads();
    for (int st = 128; st > 0; st >>= 1) {
        if (threadIdx.x < st) red[threadIdx.x] += red[threadIdx.x + st];
        __syncthreads();
    }
    float rstd = rsqrtf(red[0] / (float)D_ + EPS_);
    #pragma unroll
    for (int i = 0; i < 2; ++i) {
        int d = threadIdx.x + i * 256;
        out[(size_t)row * D_ + d] = (vals[i] - mu) * rstd * g[d] + bb[d];
    }
}

// ---------------------------------------------------------------- final head
__global__ void final_kernel(const float* __restrict__ h, const float* __restrict__ Wf,
                             const float* __restrict__ bf, float* __restrict__ out)
{
    __shared__ float red[256];
    int b = blockIdx.x;
    const float* hr = h + (size_t)(b * L_) * D_;
    float s = 0.f;
    for (int d = threadIdx.x; d < D_; d += 256) s += hr[d] * Wf[d];
    red[threadIdx.x] = s; __syncthreads();
    for (int st = 128; st > 0; st >>= 1) {
        if (threadIdx.x < st) red[threadIdx.x] += red[threadIdx.x + st];
        __syncthreads();
    }
    if (threadIdx.x == 0) out[b] = red[0] + bf[0];
}

// ================================================================ launch
extern "C" void kernel_launch(void* const* d_in, const int* in_sizes, int n_in,
                              void* d_out, int out_size, void* d_ws, size_t ws_size,
                              hipStream_t stream)
{
    (void)in_sizes; (void)n_in; (void)out_size; (void)ws_size;
    const float* x   = (const float*)d_in[0];
    const float* We  = (const float*)d_in[1];
    const float* be  = (const float*)d_in[2];
    const float* cls = (const float*)d_in[3];
    const float* Wq  = (const float*)d_in[4];
    const float* bq  = (const float*)d_in[5];
    const float* Wk  = (const float*)d_in[6];
    const float* bk  = (const float*)d_in[7];
    const float* Wv  = (const float*)d_in[8];
    const float* bv  = (const float*)d_in[9];
    const float* Wo  = (const float*)d_in[10];
    const float* bo  = (const float*)d_in[11];
    const float* g1  = (const float*)d_in[12];
    const float* b1  = (const float*)d_in[13];
    const float* g2  = (const float*)d_in[14];
    const float* b2  = (const float*)d_in[15];
    const float* W1  = (const float*)d_in[16];
    const float* bf1 = (const float*)d_in[17];
    const float* W2  = (const float*)d_in[18];
    const float* bf2 = (const float*)d_in[19];
    const float* Wf  = (const float*)d_in[20];
    const float* bfb = (const float*)d_in[21];
    float* out = (float*)d_out;

    char* wsp = (char*)d_ws;
    auto alloc = [&](size_t bytes) -> char* {
        char* p = wsp;
        wsp += (bytes + 255) & ~(size_t)255;
        return p;
    };
    float* h     = (float*)alloc(sizeof(float) * (size_t)MR_ * D_);
    float* qb    = (float*)alloc(sizeof(float) * (size_t)MR_ * D_);
    float* kb    = (float*)alloc(sizeof(float) * (size_t)MR_ * D_);
    float* vb    = (float*)alloc(sizeof(float) * (size_t)MR_ * D_);
    float* ctx   = (float*)alloc(sizeof(float) * (size_t)MR_ * D_);
    float* t1    = (float*)alloc(sizeof(float) * (size_t)MR_ * D_);
    f16*   act16 = (f16*)  alloc(sizeof(f16)   * (size_t)MR_ * D_);
    f16*   y16   = (f16*)  alloc(sizeof(f16)   * (size_t)MR_ * DFF_);
    f16*   wq16  = (f16*)  alloc(sizeof(f16)   * (size_t)NL_ * D_ * D_);
    f16*   wk16  = (f16*)  alloc(sizeof(f16)   * (size_t)NL_ * D_ * D_);
    f16*   wv16  = (f16*)  alloc(sizeof(f16)   * (size_t)NL_ * D_ * D_);
    f16*   wo16  = (f16*)  alloc(sizeof(f16)   * (size_t)NL_ * D_ * D_);
    f16*   w116  = (f16*)  alloc(sizeof(f16)   * (size_t)NL_ * DFF_ * D_);
    f16*   w216  = (f16*)  alloc(sizeof(f16)   * (size_t)NL_ * D_ * DFF_);
    float* Mbuf  = (float*)alloc(sizeof(float) * (size_t)B_ * H_ * L_);
    int*   topb  = (int*)  alloc(sizeof(int)   * (size_t)B_ * H_ * U_);
    float* vmn   = (float*)alloc(sizeof(float) * (size_t)B_ * D_);

    const dim3 blk256(256);
    auto cvt = [&](const float* src, f16* dst, int n) {
        cvt16_kernel<<<dim3((n + 255) / 256), blk256, 0, stream>>>(src, dst, n);
    };

    // weights -> f16 (both layers at once)
    cvt(Wq, wq16, NL_ * D_ * D_);
    cvt(Wk, wk16, NL_ * D_ * D_);
    cvt(Wv, wv16, NL_ * D_ * D_);
    cvt(Wo, wo16, NL_ * D_ * D_);
    cvt(W1, w116, NL_ * DFF_ * D_);
    cvt(W2, w216, NL_ * D_ * DFF_);

    // embed + cls + positional encoding
    embed_kernel<<<dim3(MR_), blk256, 0, stream>>>(x, We, be, cls, h);

    const dim3 gD(D_ / 128, MR_ / 128);     // N=512
    const dim3 gF(DFF_ / 128, MR_ / 128);   // N=2048

    for (int l = 0; l < NL_; ++l) {
        const size_t wOff = (size_t)l * D_ * D_;
        const size_t fOff = (size_t)l * DFF_ * D_;

        // Q, K, V projections
        cvt(h, act16, MR_ * D_);
        gemm_wmma_kernel<<<gD, blk256, 0, stream>>>(act16, wq16 + wOff, bq + l * D_,
                                                    qb, nullptr, MR_, D_, D_, 0);
        gemm_wmma_kernel<<<gD, blk256, 0, stream>>>(act16, wk16 + wOff, bk + l * D_,
                                                    kb, nullptr, MR_, D_, D_, 0);
        gemm_wmma_kernel<<<gD, blk256, 0, stream>>>(act16, wv16 + wOff, bv + l * D_,
                                                    vb, nullptr, MR_, D_, D_, 0);

        // prob-sparse attention
        mscore_kernel<<<dim3((B_ * H_ * L_ * 32) / 256), blk256, 0, stream>>>(qb, kb, Mbuf, l);
        topk_kernel<<<dim3(B_ * H_), blk256, 0, stream>>>(Mbuf, topb);
        vmean_kernel<<<dim3((B_ * D_ + 255) / 256), blk256, 0, stream>>>(vb, vmn);
        ctx_init_kernel<<<dim3((MR_ * D_ + 255) / 256), blk256, 0, stream>>>(vmn, ctx);
        sparse_attn_kernel<<<dim3(B_ * H_ * U_), dim3(128), 0, stream>>>(qb, kb, vb, topb, ctx);

        // output projection + residual LN
        cvt(ctx, act16, MR_ * D_);
        gemm_wmma_kernel<<<gD, blk256, 0, stream>>>(act16, wo16 + wOff, bo + l * D_,
                                                    t1, nullptr, MR_, D_, D_, 0);
        add_ln_kernel<<<dim3(MR_), blk256, 0, stream>>>(h, t1, g1 + l * D_, b1 + l * D_, h);

        // FFN
        cvt(h, act16, MR_ * D_);
        gemm_wmma_kernel<<<gF, blk256, 0, stream>>>(act16, w116 + fOff, bf1 + l * DFF_,
                                                    nullptr, y16, MR_, DFF_, D_, 1);
        gemm_wmma_kernel<<<gD, blk256, 0, stream>>>(y16, w216 + fOff, bf2 + l * D_,
                                                    t1, nullptr, MR_, D_, DFF_, 0);
        add_ln_kernel<<<dim3(MR_), blk256, 0, stream>>>(h, t1, g2 + l * D_, b2 + l * D_, h);
    }

    final_kernel<<<dim3(B_), blk256, 0, stream>>>(h, Wf, bfb, out);
}
